// MultiModalRelationGraph_48670569398799
// MI455X (gfx1250) — matrix operand order repaired and checked by reading, not compiled, not run on hardware
//
#include <hip/hip_runtime.h>
#include <hip/hip_bf16.h>
#include <stdint.h>

// ---------------------------------------------------------------------------
// Problem constants (from reference)
// ---------------------------------------------------------------------------
#define BB   16
#define TT   512
#define TAUD 1024
#define DREG 512
#define HID  256
#define HEADS 4
#define NREG 4
#define NNODES 49152          // NREG*BB*TT + BB*TAUD
#define NFACE  32768          // NREG*BB*TT
#define REGSZ  8192           // BB*TT

typedef __attribute__((ext_vector_type(16))) _Float16 v16h;
typedef __attribute__((ext_vector_type(8)))  float    v8f;

// ---------------------------------------------------------------------------
// Weight packing: f32 [K, Ncols] -> f16 in WMMA B-operand per-lane layout.
// Tile (kt, nt): lane l holds n = nt*16 + (l&15), k = kt*32 + ((l>>4)<<3) + (e&7) + ((e>>3)<<4),
// e = 0..15 (table order).  out index = ((kt*ntiles + nt)*32 + lane)*16 + e
// ---------------------------------------------------------------------------
__global__ __launch_bounds__(256) void pack_w_kernel(
    const float* __restrict__ W, _Float16* __restrict__ out,
    int K, int Ncols, int total)
{
  int tid = blockIdx.x * 256 + threadIdx.x;
  if (tid >= total) return;
  int e    = tid & 15;
  int lane = (tid >> 4) & 31;
  int rest = tid >> 9;
  int ntiles = Ncols >> 4;
  int nt = rest % ntiles;
  int kt = rest / ntiles;
  int n = nt * 16 + (lane & 15);
  int k = kt * 32 + ((lane >> 4) << 3) + (e & 7) + ((e >> 3) << 4);
  out[tid] = (_Float16)W[(size_t)k * Ncols + n];
}

// ---------------------------------------------------------------------------
// Projection GEMM (A is f32 from the harness inputs):
// Ch[M,Ncols](f16) = A[M,K](f32 -> f16) @ Bpack + bias
// grid = (M/16, Ncols/256), block = 128 (4 waves); wave covers 64 cols.
// ---------------------------------------------------------------------------
__global__ __launch_bounds__(128) void gemm_wmma_f32a_kernel(
    const float* __restrict__ A, const _Float16* __restrict__ Bp,
    const float* __restrict__ bias, _Float16* __restrict__ Ch,
    int K, int Ncols)
{
  __shared__ uint32_t As32[16][16];   // 16 rows x 32 halves (packed u32)

  const int tid  = threadIdx.x;
  const int lane = tid & 31;
  const int wave = tid >> 5;
  const int row0 = blockIdx.x * 16;
  const int wcol = blockIdx.y * 256 + wave * 64;
  const int ntiles = Ncols >> 4;
  const int m    = lane & 15;
  const int hi8  = (lane >> 4) << 3;
  const int hi4  = (lane >> 4) << 2;

  const int r  = tid >> 3;            // staging row 0..15
  const int kk = (tid & 7) << 2;      // staging k 0,4,...,28

  v8f acc[4];
  acc[0] = v8f{}; acc[1] = v8f{}; acc[2] = v8f{}; acc[3] = v8f{};

  const int nkt = K >> 5;
  for (int kt = 0; kt < nkt; ++kt) {
    const float4 f = *(const float4*)(A + (size_t)(row0 + r) * K + kt * 32 + kk);
    if (kt + 1 < nkt)
      __builtin_prefetch(A + (size_t)(row0 + r) * K + (kt + 1) * 32 + kk, 0, 1);
    union { _Float16 h[2]; uint32_t u; } p0, p1;
    p0.h[0] = (_Float16)f.x; p0.h[1] = (_Float16)f.y;
    p1.h[0] = (_Float16)f.z; p1.h[1] = (_Float16)f.w;
    __syncthreads();
    As32[r][(kk >> 1)]     = p0.u;
    As32[r][(kk >> 1) + 1] = p1.u;
    __syncthreads();

    union { v16h v; uint32_t u[8]; } afrag;
#pragma unroll
    for (int v = 0; v < 8; ++v)
      afrag.u[v] = As32[m][(v < 4 ? v : v + 4) + hi4];

#pragma unroll
    for (int t = 0; t < 4; ++t) {
      int nt = (wcol >> 4) + t;
      const v16h bfrag =
          *(const v16h*)(Bp + ((size_t)kt * ntiles + nt) * 512 + lane * 16);
      acc[t] = __builtin_amdgcn_wmma_f32_16x16x32_f16(
          false, afrag.v, false, bfrag, (short)0, acc[t], false, false);
    }
  }

#pragma unroll
  for (int t = 0; t < 4; ++t) {
    int col = wcol + t * 16 + m;
    float bv = bias[col];
#pragma unroll
    for (int i = 0; i < 8; ++i) {
      int rr = row0 + hi8 + i;
      Ch[(size_t)rr * Ncols + col] = (_Float16)(acc[t][i] + bv);
    }
  }
}

// ---------------------------------------------------------------------------
// Layer GEMM (A is pre-converted f16): C[M,Ncols](f32) = Ah @ Bpack.
// A tile staged via GLOBAL_LOAD_ASYNC_TO_LDS_B128: waves 0-1 each issue one
// per-lane async b128 (512B each -> 1KB tile), wait own ASYNCcnt, barrier.
// ---------------------------------------------------------------------------
__global__ __launch_bounds__(128) void gemm_wmma_f16a_kernel(
    const _Float16* __restrict__ Ah, const _Float16* __restrict__ Bp,
    float* __restrict__ C, int K, int Ncols)
{
  __shared__ uint32_t As32[16][16];   // raw row-major 16x32-half tile

  const int tid  = threadIdx.x;
  const int lane = tid & 31;
  const int wave = tid >> 5;
  const int row0 = blockIdx.x * 16;
  const int wcol = blockIdx.y * 256 + wave * 64;
  const int ntiles = Ncols >> 4;
  const int m    = lane & 15;
  const int hi8  = (lane >> 4) << 3;
  const int hi4  = (lane >> 4) << 2;

  // async copy mapping (waves 0,1): chunk q -> row ar, half-offset ap (16B)
  const int q  = wave * 32 + lane;    // 0..63 meaningful
  const int ar = q >> 2;              // 0..15
  const int ap = (q & 3) << 3;        // 0,8,16,24 halves
  const unsigned ldsoff =
      (unsigned)(uintptr_t)(__attribute__((address_space(3))) uint32_t*)&As32[0][0]
      + (unsigned)(ar * 32 + ap) * 2u;

  v8f acc[4];
  acc[0] = v8f{}; acc[1] = v8f{}; acc[2] = v8f{}; acc[3] = v8f{};

  const int nkt = K >> 5;
  for (int kt = 0; kt < nkt; ++kt) {
    __syncthreads();                       // previous tile fully consumed
    if (wave < 2) {
      const _Float16* gsrc = Ah + (size_t)(row0 + ar) * K + kt * 32 + ap;
      asm volatile("global_load_async_to_lds_b128 %0, %1, off"
                   :: "v"(ldsoff), "v"((uint64_t)(uintptr_t)gsrc)
                   : "memory");
      asm volatile("s_wait_asynccnt 0x0" ::: "memory");
    }
    __syncthreads();                       // tile visible to all waves

    union { v16h v; uint32_t u[8]; } afrag;
#pragma unroll
    for (int v = 0; v < 8; ++v)
      afrag.u[v] = As32[m][(v < 4 ? v : v + 4) + hi4];

    if (kt + 1 < nkt)
      __builtin_prefetch(
          Bp + ((size_t)(kt + 1) * ntiles + (wcol >> 4)) * 512 + lane * 16, 0, 1);

#pragma unroll
    for (int t = 0; t < 4; ++t) {
      int nt = (wcol >> 4) + t;
      const v16h bfrag =
          *(const v16h*)(Bp + ((size_t)kt * ntiles + nt) * 512 + lane * 16);
      acc[t] = __builtin_amdgcn_wmma_f32_16x16x32_f16(
          false, afrag.v, false, bfrag, (short)0, acc[t], false, false);
    }
  }

#pragma unroll
  for (int t = 0; t < 4; ++t) {
    int col = wcol + t * 16 + m;
#pragma unroll
    for (int i = 0; i < 8; ++i) {
      int rr = row0 + hi8 + i;
      C[(size_t)rr * Ncols + col] = acc[t][i];
    }
  }
}

// ---------------------------------------------------------------------------
// alpha_s[n,h] = sum_c h[n,h,c]*a_s[h,c]; alpha_d analogous. Wave per node.
// ---------------------------------------------------------------------------
__global__ __launch_bounds__(256) void alpha_kernel(
    const float* __restrict__ h, const float* __restrict__ a_s,
    const float* __restrict__ a_d, float* __restrict__ alpS,
    float* __restrict__ alpD)
{
  int n    = blockIdx.x * 8 + (threadIdx.x >> 5);
  int lane = threadIdx.x & 31;
  const float* hr = h + (size_t)n * 1024;
#pragma unroll
  for (int hh = 0; hh < HEADS; ++hh) {
    float ss = 0.0f, sd = 0.0f;
    for (int c = lane; c < HID; c += 32) {
      float v = hr[hh * HID + c];
      ss += v * a_s[hh * HID + c];
      sd += v * a_d[hh * HID + c];
    }
#pragma unroll
    for (int off = 16; off; off >>= 1) {
      ss += __shfl_xor(ss, off, 32);
      sd += __shfl_xor(sd, off, 32);
    }
    if (lane == 0) { alpS[n * 4 + hh] = ss; alpD[n * 4 + hh] = sd; }
  }
}

// ---------------------------------------------------------------------------
// Closed-form in-edge enumeration (max in-degree 5, incl. self loop).
// ---------------------------------------------------------------------------
__device__ __forceinline__ int build_srcs(int n, int* s) {
  int ns = 0;
  if (n < NFACE) {
    int i   = n >> 13;        // region
    int rem = n & 8191;       // b*T + t
    int t   = rem & 511;
#pragma unroll
    for (int j = 0; j < NREG; ++j)
      if (j != i) s[ns++] = (j << 13) + rem;       // cross-region, same (b,t)
    if (t > 0) s[ns++] = n - 1;                    // temporal t-1
    s[ns++] = n;                                   // self loop
  } else {
    int a  = n - NFACE;
    int ta = a & 1023;
    if ((ta & 1) == 0) {                           // t_aud = 2t hits only even
      int b = a >> 10;
      int t = ta >> 1;
      s[ns++] = (1 << 13) + b * TT + t;            // left_eye
      s[ns++] = (2 << 13) + b * TT + t;            // right_eye
    }
    s[ns++] = n;                                   // self loop
  }
  return ns;
}

__device__ __forceinline__ void edge_softmax(
    const float* __restrict__ alpS, float ad, const int* s, int ns,
    int hh, float* w)
{
  float m = -1e30f;
#pragma unroll 5
  for (int k = 0; k < ns; ++k) {
    float v = alpS[s[k] * 4 + hh] + ad;
    v = v > 0.0f ? v : 0.2f * v;          // leaky_relu(0.2)
    w[k] = v;
    m = fmaxf(m, v);
  }
  float den = 0.0f;
#pragma unroll 5
  for (int k = 0; k < ns; ++k) { w[k] = __expf(w[k] - m); den += w[k]; }
  den += 1e-16f;
  float inv = 1.0f / den;
#pragma unroll 5
  for (int k = 0; k < ns; ++k) w[k] *= inv;
}

// Gather + softmax + bias (concat heads): out is f16 [N,1024] (next GEMM's A)
__global__ __launch_bounds__(256) void gather_concat_kernel(
    const float* __restrict__ h, const float* __restrict__ alpS,
    const float* __restrict__ alpD, const float* __restrict__ bias,
    _Float16* __restrict__ out)
{
  int n = blockIdx.x;
  int s[5];
  int ns = build_srcs(n, s);
  int c = threadIdx.x;
#pragma unroll
  for (int hh = 0; hh < HEADS; ++hh) {
    float w[5];
    edge_softmax(alpS, alpD[n * 4 + hh], s, ns, hh, w);
    float acc = 0.0f;
#pragma unroll 5
    for (int k = 0; k < ns; ++k)
      acc += w[k] * h[(size_t)s[k] * 1024 + hh * HID + c];
    out[(size_t)n * 1024 + hh * HID + c] =
        (_Float16)(acc + bias[hh * HID + c]);
  }
}

// Gather + softmax + head-mean (layer 2): out f32 [N, 256] (LayerNorm input)
__global__ __launch_bounds__(256) void gather_mean_kernel(
    const float* __restrict__ h, const float* __restrict__ alpS,
    const float* __restrict__ alpD, const float* __restrict__ bias,
    float* __restrict__ out)
{
  int n = blockIdx.x;
  int s[5];
  int ns = build_srcs(n, s);
  int c = threadIdx.x;
  float acc = 0.0f;
#pragma unroll
  for (int hh = 0; hh < HEADS; ++hh) {
    float w[5];
    edge_softmax(alpS, alpD[n * 4 + hh], s, ns, hh, w);
#pragma unroll 5
    for (int k = 0; k < ns; ++k)
      acc += w[k] * h[(size_t)s[k] * 1024 + hh * HID + c];
  }
  out[(size_t)n * HID + c] = acc * 0.25f + bias[c];
}

// LayerNorm over HID per node
__global__ __launch_bounds__(256) void ln_kernel(
    const float* __restrict__ x, const float* __restrict__ g,
    const float* __restrict__ b, float* __restrict__ y)
{
  __shared__ float red[256];
  int n = blockIdx.x, c = threadIdx.x;
  float v = x[(size_t)n * HID + c];
  red[c] = v; __syncthreads();
#pragma unroll
  for (int off = 128; off; off >>= 1) {
    if (c < off) red[c] += red[c + off];
    __syncthreads();
  }
  float mu = red[0] * (1.0f / HID);
  __syncthreads();
  float d = v - mu;
  red[c] = d * d; __syncthreads();
#pragma unroll
  for (int off = 128; off; off >>= 1) {
    if (c < off) red[c] += red[c + off];
    __syncthreads();
  }
  float var = red[0] * (1.0f / HID);
  y[(size_t)n * HID + c] = d * rsqrtf(var + 1e-5f) * g[c] + b[c];
}

// Final mean over groups of 3072 nodes -> [16, 256]
__global__ __launch_bounds__(256) void reduce_kernel(
    const float* __restrict__ y, float* __restrict__ out)
{
  int g = blockIdx.x, c = threadIdx.x;
  const float* p = y + (size_t)g * 3072 * HID + c;
  float s = 0.0f;
  for (int i = 0; i < 3072; ++i) s += p[(size_t)i * HID];
  out[g * HID + c] = s * (1.0f / 3072.0f);
}

// ---------------------------------------------------------------------------
// Host launch
// ---------------------------------------------------------------------------
extern "C" void kernel_launch(void* const* d_in, const int* in_sizes, int n_in,
                              void* d_out, int out_size, void* d_ws, size_t ws_size,
                              hipStream_t stream) {
  (void)in_sizes; (void)n_in; (void)out_size; (void)ws_size;
  const float* regions[4] = {(const float*)d_in[0], (const float*)d_in[1],
                             (const float*)d_in[2], (const float*)d_in[3]};
  const float* audio   = (const float*)d_in[4];
  const float* wreg[5] = {(const float*)d_in[5], (const float*)d_in[7],
                          (const float*)d_in[9], (const float*)d_in[11],
                          (const float*)d_in[13]};
  const float* breg[5] = {(const float*)d_in[6], (const float*)d_in[8],
                          (const float*)d_in[10], (const float*)d_in[12],
                          (const float*)d_in[14]};
  const float* Wl[3]   = {(const float*)d_in[15], (const float*)d_in[19],
                          (const float*)d_in[23]};
  const float* asl[3]  = {(const float*)d_in[16], (const float*)d_in[20],
                          (const float*)d_in[24]};
  const float* adl[3]  = {(const float*)d_in[17], (const float*)d_in[21],
                          (const float*)d_in[25]};
  const float* bl[3]   = {(const float*)d_in[18], (const float*)d_in[22],
                          (const float*)d_in[26]};
  const float* ln_g = (const float*)d_in[27];
  const float* ln_b = (const float*)d_in[28];
  // d_in[29]=src, d_in[30]=dst unused: edge structure is closed-form.
  float* out = (float*)d_out;

  // workspace carve-out (256B aligned)
  char* ws = (char*)d_ws;
  size_t off = 0;
  auto take = [&](size_t bytes) -> char* {
    char* p = ws + off;
    off = (off + bytes + 255) & ~(size_t)255;
    return p;
  };
  float*    bufH  = (float*)take((size_t)NNODES * 1024 * sizeof(float));
  _Float16* bufXh = (_Float16*)take((size_t)NNODES * 1024 * sizeof(_Float16));
  float*    bufX3 = (float*)take((size_t)NNODES * HID * sizeof(float));
  float*    alpS  = (float*)take((size_t)NNODES * 4 * sizeof(float));
  float*    alpD  = (float*)take((size_t)NNODES * 4 * sizeof(float));
  _Float16* pw[5];
  for (int i = 0; i < 5; ++i)
    pw[i] = (_Float16*)take((size_t)DREG * HID * sizeof(_Float16));
  _Float16* pW[3];
  pW[0] = (_Float16*)take((size_t)HID * 1024 * sizeof(_Float16));
  pW[1] = (_Float16*)take((size_t)1024 * 1024 * sizeof(_Float16));
  pW[2] = (_Float16*)take((size_t)1024 * 1024 * sizeof(_Float16));

  // ---- 1. pack weights into WMMA B-operand f16 layout ----
  for (int i = 0; i < 5; ++i) {
    int total = DREG * HID;
    pack_w_kernel<<<(total + 255) / 256, 256, 0, stream>>>(
        wreg[i], pw[i], DREG, HID, total);
  }
  {
    int t0 = HID * 1024;
    pack_w_kernel<<<(t0 + 255) / 256, 256, 0, stream>>>(Wl[0], pW[0], HID, 1024, t0);
    int t1 = 1024 * 1024;
    pack_w_kernel<<<(t1 + 255) / 256, 256, 0, stream>>>(Wl[1], pW[1], 1024, 1024, t1);
    pack_w_kernel<<<(t1 + 255) / 256, 256, 0, stream>>>(Wl[2], pW[2], 1024, 1024, t1);
  }

  // ---- 2. input projections -> x0 (bufXh, f16, width 256) ----
  for (int r = 0; r < 4; ++r) {
    gemm_wmma_f32a_kernel<<<dim3(REGSZ / 16, 1), 128, 0, stream>>>(
        regions[r], pw[r], breg[r], bufXh + (size_t)r * REGSZ * HID, DREG, HID);
  }
  gemm_wmma_f32a_kernel<<<dim3((BB * TAUD) / 16, 1), 128, 0, stream>>>(
      audio, pw[4], breg[4], bufXh + (size_t)NFACE * HID, DREG, HID);

  // ---- 3. GAT layer 0: x0(256,f16) -> h0(1024,f32) -> x1(1024,f16) ----
  gemm_wmma_f16a_kernel<<<dim3(NNODES / 16, 4), 128, 0, stream>>>(
      bufXh, pW[0], bufH, HID, 1024);
  alpha_kernel<<<NNODES / 8, 256, 0, stream>>>(bufH, asl[0], adl[0], alpS, alpD);
  gather_concat_kernel<<<NNODES, 256, 0, stream>>>(bufH, alpS, alpD, bl[0], bufXh);

  // ---- 4. GAT layer 1 ----
  gemm_wmma_f16a_kernel<<<dim3(NNODES / 16, 4), 128, 0, stream>>>(
      bufXh, pW[1], bufH, 1024, 1024);
  alpha_kernel<<<NNODES / 8, 256, 0, stream>>>(bufH, asl[1], adl[1], alpS, alpD);
  gather_concat_kernel<<<NNODES, 256, 0, stream>>>(bufH, alpS, alpD, bl[1], bufXh);

  // ---- 5. GAT layer 2 (head mean) -> x3 (f32, width 256) ----
  gemm_wmma_f16a_kernel<<<dim3(NNODES / 16, 4), 128, 0, stream>>>(
      bufXh, pW[2], bufH, 1024, 1024);
  alpha_kernel<<<NNODES / 8, 256, 0, stream>>>(bufH, asl[2], adl[2], alpS, alpD);
  gather_mean_kernel<<<NNODES, 256, 0, stream>>>(bufH, alpS, alpD, bl[2], bufX3);

  // ---- 6. LayerNorm + group mean -> [16, 256] ----
  ln_kernel<<<NNODES, 256, 0, stream>>>(bufX3, ln_g, ln_b, bufH);
  reduce_kernel<<<BB, 256, 0, stream>>>(bufH, out);
}
